// Attention_6511170421658
// MI455X (gfx1250) — matrix operand order repaired and checked by reading, not compiled, not run on hardware
//
#include <hip/hip_runtime.h>
#include <hip/hip_bf16.h>

typedef __attribute__((ext_vector_type(16))) _Float16 v16h;
typedef __attribute__((ext_vector_type(8)))  float    v8f;

#define B_SZ   8
#define M_SZ   512
#define D_SZ   64
#define ROWS   16            // i-rows per workgroup
#define JCHUNK 64            // j rows staged per LDS chunk
#define NCHUNK (M_SZ / JCHUNK)
#define KT_PITCH 68          // floats per row: 272 B = 16-B aligned, b128-legal
#define WA_PITCH 520         // M_SZ + 8 pad (halfs)

// Low 32 bits of a flat pointer into the LDS aperture are the raw LDS byte
// offset (ISA 10.2).
__device__ __forceinline__ uint32_t lds_off(const void* p) {
    return (uint32_t)(uintptr_t)p;
}

// Issue this thread's share (4 x b128) of one 64x64-f32 k chunk as
// CDNA5 async global->LDS copies (tracked by ASYNCcnt, no VGPR round-trip).
__device__ __forceinline__ void stage_chunk_async(const float* __restrict__ src,
                                                  uint32_t lds_base, int tid) {
    #pragma unroll
    for (int it = 0; it < 4; ++it) {
        int f  = tid + it * 256;                 // float4 index in chunk [0,1024)
        int j  = f >> 4, dq = f & 15;            // row / float4-within-row
        uint64_t ga = (uint64_t)(uintptr_t)src + (uint64_t)f * 16u;
        uint32_t la = lds_base + (uint32_t)(j * (KT_PITCH * 4) + dq * 16);
        asm volatile("global_load_async_to_lds_b128 %0, %1, off"
                     :: "v"(la), "v"(ga) : "memory");
    }
}

__global__ __launch_bounds__(256)
void laplace_attn_kernel(const float* __restrict__ K,
                         const float* __restrict__ V,
                         float* __restrict__ out)
{
    // kt (pass A) is dead before wA (pass B/C) is born -> union
    __shared__ union __align__(16) {
        float    kt[2][JCHUNK * KT_PITCH];   // double-buffered k chunk, row-major
        _Float16 wA[ROWS * WA_PITCH];        // softmax weights (f16) = WMMA A
    } su;
    __shared__ float s_scores[ROWS * M_SZ];            // 32 KB
    __shared__ __align__(16) float s_vi[ROWS * D_SZ];  // 4 KB

    const int tid  = threadIdx.x;
    const int lane = tid & 31;
    const int wave = tid >> 5;

    const int b  = blockIdx.x >> 5;           // 32 i-tiles per batch
    const int i0 = (blockIdx.x & 31) * ROWS;

    const float* Kb = K + (size_t)b * M_SZ * D_SZ;
    const float* Vb = V + (size_t)b * M_SZ * D_SZ;

    const uint32_t ktb[2] = { lds_off(&su.kt[0][0]), lds_off(&su.kt[1][0]) };

    // kick off chunk 0 immediately; prefetch V (pass-C operand) into cache
    stage_chunk_async(Kb, ktb[0], tid);
    __builtin_prefetch(Vb + (size_t)tid * 128, 0, 1);   // global_prefetch_b8

    // stage v[i-tile] (the "query side" of the L1 distance)
    for (int e = tid; e < ROWS * D_SZ; e += 256)
        s_vi[e] = Vb[(size_t)i0 * D_SZ + e];

    // ---- Pass A: L1-distance scores, k streamed via async double-buffer ----
    const int r0 = wave * 2;                  // each wave owns 2 i-rows
    for (int c = 0; c < NCHUNK; ++c) {
        if (c + 1 < NCHUNK) {
            stage_chunk_async(Kb + (size_t)(c + 1) * JCHUNK * D_SZ,
                              ktb[(c + 1) & 1], tid);
            // 4 new ops outstanding; async loads complete in order ->
            // <=4 outstanding means chunk c has fully landed in LDS
            asm volatile("s_wait_asynccnt 0x4" ::: "memory");
        } else {
            asm volatile("s_wait_asynccnt 0x0" ::: "memory");
        }
        __syncthreads();                      // chunk c visible to all waves
        const float* kc = &su.kt[c & 1][0];
        #pragma unroll
        for (int rr = 0; rr < 2; ++rr) {
            const int r = r0 + rr;
            float4 vreg[16];                  // broadcast row v[i], hoisted
            #pragma unroll
            for (int dq = 0; dq < 16; ++dq)
                vreg[dq] = *(const float4*)&s_vi[r * D_SZ + dq * 4];
            #pragma unroll
            for (int jg = 0; jg < JCHUNK / 32; ++jg) {
                const int jl = jg * 32 + lane;
                const float4* kr = (const float4*)&kc[jl * KT_PITCH];
                float acc = 0.f;
                #pragma unroll
                for (int dq = 0; dq < 16; ++dq) {
                    float4 kv = kr[dq];       // ds_load_b128
                    acc += fabsf(kv.x - vreg[dq].x) + fabsf(kv.y - vreg[dq].y)
                         + fabsf(kv.z - vreg[dq].z) + fabsf(kv.w - vreg[dq].w);
                }
                s_scores[r * M_SZ + c * JCHUNK + jl] = 0.5f * acc;
            }
        }
        __syncthreads();                      // buffer consumed before refill
    }

    // ---- Pass B: softmax along j (wave32 shuffle reductions) ----
    #pragma unroll
    for (int rr = 0; rr < 2; ++rr) {
        const int r = r0 + rr;
        float m = -__builtin_inff();
        for (int t = 0; t < M_SZ / 32; ++t)
            m = fmaxf(m, s_scores[r * M_SZ + t * 32 + lane]);
        for (int off = 16; off > 0; off >>= 1)
            m = fmaxf(m, __shfl_xor(m, off, 32));
        float sum = 0.f;
        for (int t = 0; t < M_SZ / 32; ++t) {
            int idx = r * M_SZ + t * 32 + lane;
            float e = __expf(s_scores[idx] - m);
            s_scores[idx] = e;
            sum += e;
        }
        for (int off = 16; off > 0; off >>= 1)
            sum += __shfl_xor(sum, off, 32);
        float inv = 1.0f / sum;
        for (int t = 0; t < M_SZ / 32; ++t)
            su.wA[r * WA_PITCH + t * 32 + lane] =
                (_Float16)(s_scores[r * M_SZ + t * 32 + lane] * inv);
    }
    __syncthreads();   // all 16 rows of wA visible to matmul waves

    // ---- Pass C: rep[16x64] = W[16x512] x V[512x64], v_wmma_f32_16x16x32_f16
    if (wave < 4) {
        const int n0 = wave * 16;             // one 16-wide n-tile per wave
        const int n  = lane & 15;
        const int kg = lane >> 4;             // lane half-group
        v8f acc = {};
        #pragma unroll 2
        for (int ks = 0; ks < M_SZ / 32; ++ks) {
            const int k0 = ks * 32;
            v16h a, bm;
            #pragma unroll
            for (int vv = 0; vv < 8; ++vv) {
                // A 16x32 f16 layout (ISA 7.12.2): VGPR v<4 -> K=2v, v>=4 -> 16+2(v-4);
                // lanes 16-31 offset K by +8
                int ak = k0 + ((vv < 4) ? (2 * vv) : (16 + 2 * (vv - 4))) + 8 * kg;
                a[2 * vv]     = su.wA[(lane & 15) * WA_PITCH + ak];
                a[2 * vv + 1] = su.wA[(lane & 15) * WA_PITCH + ak + 1];
                // B 32x16 f16 layout: same K order as A (v<4 -> K=2v, v>=4 -> 16+2(v-4); +8 for lanes 16-31)
                int bk = k0 + ((vv < 4) ? (2 * vv) : (16 + 2 * (vv - 4))) + 8 * kg;
                bm[2 * vv]     = (_Float16)Vb[(size_t)bk       * D_SZ + n0 + n];
                bm[2 * vv + 1] = (_Float16)Vb[(size_t)(bk + 1) * D_SZ + n0 + n];
            }
            acc = __builtin_amdgcn_wmma_f32_16x16x32_f16(
                      /*neg_a=*/false, a, /*neg_b=*/false, bm,
                      /*c_mod=*/(short)0, acc, /*reuse_a=*/false, /*reuse_b=*/false);
        }
        // C/D layout: VGPR r -> M = r + 8*(lane>=16), N = lane%16
        float* ob = out + ((size_t)b * M_SZ + i0) * D_SZ;
        #pragma unroll
        for (int rr = 0; rr < 8; ++rr) {
            int mrow = rr + 8 * kg;
            ob[(size_t)mrow * D_SZ + n0 + n] = acc[rr];
        }
    }
}

extern "C" void kernel_launch(void* const* d_in, const int* in_sizes, int n_in,
                              void* d_out, int out_size, void* d_ws, size_t ws_size,
                              hipStream_t stream) {
    (void)in_sizes; (void)n_in; (void)out_size; (void)d_ws; (void)ws_size;
    const float* k = (const float*)d_in[0];
    const float* v = (const float*)d_in[1];
    // q (d_in[2]) is unused by the reference computation
    float* out = (float*)d_out;
    dim3 grid(B_SZ * (M_SZ / ROWS));   // 8 * 32 = 256 workgroups
    dim3 block(256);                   // 8 wave32 waves
    laplace_attn_kernel<<<grid, block, 0, stream>>>(k, v, out);
}
